// KanGNN_17540646437274
// MI455X (gfx1250) — matrix operand order repaired and checked by reading, not compiled, not run on hardware
//
#include <hip/hip_runtime.h>

// ---------------- problem constants (match reference) ----------------
#define NN   100000      // nodes
#define FF   165         // input features
#define HH   64          // hidden
#define OO   2           // outputs
#define GG   8           // fourier grid
#define EE   1600000     // edges
#define NKAN 2

#define NTILES (NN / 16)           // 6250 16-node tiles
#define WAVES_PER_BLK 8

typedef _Float16 half16 __attribute__((ext_vector_type(16)));
typedef float    float8 __attribute__((ext_vector_type(8)));

__device__ __forceinline__ float8 wmma_f16(half16 a, half16 b, float8 c) {
    // D = A(16x32 f16) * B(32x16 f16) + C(16x16 f32)
    return __builtin_amdgcn_wmma_f32_16x16x32_f16(
        /*neg_a=*/false, a, /*neg_b=*/false, b,
        /*c_mod=*/(short)0, c, /*reuse_a=*/false, /*reuse_b=*/false);
}

// ---------------------------------------------------------------------
// Repack W_in [64][165] f32 -> B-fragment f16 buffer:
// Wb[((ct*6 + chunk)*32 + lane)*16 + j],  K = chunk*32 + ((lane&16)>>1) + (j&7) + ((j&8)<<1),
// element = W_in[o=ct*16+(lane&15)][K], zero-padded for K >= 165.
// ---------------------------------------------------------------------
__global__ void prep_win(const float* __restrict__ Win, _Float16* __restrict__ Wb) {
    int t = blockIdx.x * blockDim.x + threadIdx.x;
    if (t >= 4 * 6 * 32 * 16) return;
    int j     = t & 15;
    int lane  = (t >> 4) & 31;
    int chunk = (t >> 9) % 6;
    int ct    = t / (6 * 32 * 16);
    int K = chunk * 32 + ((lane & 16) >> 1) + (j & 7) + ((j & 8) << 1);
    int o = ct * 16 + (lane & 15);
    float v = (K < FF) ? Win[o * FF + K] : 0.f;
    Wb[t] = (_Float16)v;
}

// ---------------------------------------------------------------------
// Repack KAN coeffs [2][2][64][64][8] f32 -> B-fragment f16 buffer.
// K-index convention inside a 32-chunk: kin = i_local*16 + g*2 + trig,
// i = chunk*2 + i_local.  Bp[((((l*4+ct)*32+chunk)*32+lane)*16 + j]
// ---------------------------------------------------------------------
__global__ void prep_bp(const float* __restrict__ coeffs, _Float16* __restrict__ Bp) {
    int t = blockIdx.x * blockDim.x + threadIdx.x;
    if (t >= NKAN * 4 * 32 * 32 * 16) return;
    int j     = t & 15;
    int lane  = (t >> 4)  & 31;
    int chunk = (t >> 9)  & 31;
    int ct    = (t >> 14) & 3;
    int l     = t >> 16;
    int kin    = ((lane & 16) >> 1) + (j & 7) + ((j & 8) << 1);          // 0..31 within chunk
    int ilocal = kin >> 4;
    int g      = (kin & 15) >> 1;
    int trig   = kin & 1;                  // 0=cos, 1=sin
    int i      = chunk * 2 + ilocal;
    int o      = ct * 16 + (lane & 15);
    float v = coeffs[(((l * 2 + trig) * HH + o) * HH + i) * GG + g];
    Bp[t] = (_Float16)v;
}

// ---------------------------------------------------------------------
// lin_in:  h[N][64] = x[N][165] @ W_in^T   via v_wmma_f32_16x16x32_f16
// one wave = one 16-node tile, all 4 output column tiles.
// ---------------------------------------------------------------------
__global__ void __launch_bounds__(256) lin_in_kernel(const float* __restrict__ x,
                                                     const _Float16* __restrict__ Wb,
                                                     float* __restrict__ h) {
    const int wave = threadIdx.x >> 5;
    const int lane = threadIdx.x & 31;
    const int tile = blockIdx.x * WAVES_PER_BLK + wave;
    if (tile >= NTILES) return;                 // whole-wave exit: EXEC stays all-1
    const int m    = lane & 15;
    const int row  = tile * 16 + m;
    const int koff = (lane & 16) >> 1;          // 0 or 8
    float8 acc0 = {}, acc1 = {}, acc2 = {}, acc3 = {};
    for (int c = 0; c < 6; ++c) {
        const int kbase = c * 32 + koff;
        half16 a;
#pragma unroll
        for (int j = 0; j < 8; ++j) {
            int ka = kbase + j;
            int kb = kbase + 16 + j;
            a[j]     = (ka < FF) ? (_Float16)x[row * FF + ka] : (_Float16)0.f;
            a[8 + j] = (kb < FF) ? (_Float16)x[row * FF + kb] : (_Float16)0.f;
        }
        half16 b0 = *(const half16*)(Wb + (((0 * 6 + c) * 32 + lane) << 4));
        half16 b1 = *(const half16*)(Wb + (((1 * 6 + c) * 32 + lane) << 4));
        half16 b2 = *(const half16*)(Wb + (((2 * 6 + c) * 32 + lane) << 4));
        half16 b3 = *(const half16*)(Wb + (((3 * 6 + c) * 32 + lane) << 4));
        acc0 = wmma_f16(a, b0, acc0);
        acc1 = wmma_f16(a, b1, acc1);
        acc2 = wmma_f16(a, b2, acc2);
        acc3 = wmma_f16(a, b3, acc3);
    }
    const int roff = (lane & 16) >> 1;          // 0 or 8
#pragma unroll
    for (int j = 0; j < 8; ++j) {
        int r = tile * 16 + roff + j;
        h[r * HH +  0 + m] = acc0[j];
        h[r * HH + 16 + m] = acc1[j];
        h[r * HH + 32 + m] = acc2[j];
        h[r * HH + 48 + m] = acc3[j];
    }
}

// ---------------------------------------------------------------------
// SpMM: hout[row] += w * hin[col]; 64 threads per edge (coalesced),
// global f32 atomics; h tables are L2-resident (25.6 MB << 192 MB).
// ---------------------------------------------------------------------
__global__ void __launch_bounds__(256) spmm_kernel(const int*   __restrict__ ei,
                                                   const float* __restrict__ ew,
                                                   const float* __restrict__ hin,
                                                   float*       __restrict__ hout) {
    unsigned t = blockIdx.x * 256u + threadIdx.x;
    if (t >= (unsigned)EE * HH) return;
    int e = (int)(t >> 6);
    int f = (int)(t & 63);
    int r = ei[e];             // dst
    int c = ei[EE + e];        // src
    float w = ew[e];
    atomicAdd(&hout[r * HH + f], hin[c * HH + f] * w);
}

// ---------------------------------------------------------------------
// Fourier-KAN layer: per 16-node tile, synthesize A = [cos|sin](s*k) in
// f16 registers from an LDS-staged tile, multiply against pre-packed
// coefficient B fragments.  K total = 64*8*2 = 1024 -> 32 chunks of 32.
// ---------------------------------------------------------------------
__global__ void __launch_bounds__(256) kan_kernel(const float* __restrict__ s_in,
                                                  const _Float16* __restrict__ Bp,
                                                  float* __restrict__ hout) {
    __shared__ float st[WAVES_PER_BLK][16][65];   // pad 64->65: bank-conflict-free
    const int wave = threadIdx.x >> 5;
    const int lane = threadIdx.x & 31;
    const int tile = blockIdx.x * WAVES_PER_BLK + wave;
    const bool active = (tile < NTILES);
    if (active) {
        for (int idx = lane; idx < 16 * 64; idx += 32) {
            int r = idx >> 6, cc = idx & 63;
            st[wave][r][cc] = s_in[(tile * 16 + r) * HH + cc];
        }
    }
    __syncthreads();
    if (!active) return;
    const int m     = lane & 15;
    const int ghalf = (lane & 16) >> 2;           // 0 or 4: which 4 grid freqs
    float8 acc0 = {}, acc1 = {}, acc2 = {}, acc3 = {};
    for (int c = 0; c < 32; ++c) {
        float s0 = st[wave][m][2 * c];
        float s1 = st[wave][m][2 * c + 1];
        half16 a;
#pragma unroll
        for (int gg = 0; gg < 4; ++gg) {
            float kmul = (float)(ghalf + gg + 1);
            float a0 = s0 * kmul, a1 = s1 * kmul;
            a[gg * 2]         = (_Float16)__cosf(a0);   // trig=0
            a[gg * 2 + 1]     = (_Float16)__sinf(a0);   // trig=1
            a[8 + gg * 2]     = (_Float16)__cosf(a1);
            a[8 + gg * 2 + 1] = (_Float16)__sinf(a1);
        }
        half16 b0 = *(const half16*)(Bp + (((0 * 32 + c) * 32 + lane) << 4));
        half16 b1 = *(const half16*)(Bp + (((1 * 32 + c) * 32 + lane) << 4));
        half16 b2 = *(const half16*)(Bp + (((2 * 32 + c) * 32 + lane) << 4));
        half16 b3 = *(const half16*)(Bp + (((3 * 32 + c) * 32 + lane) << 4));
        acc0 = wmma_f16(a, b0, acc0);
        acc1 = wmma_f16(a, b1, acc1);
        acc2 = wmma_f16(a, b2, acc2);
        acc3 = wmma_f16(a, b3, acc3);
    }
    const int roff = (lane & 16) >> 1;
#pragma unroll
    for (int j = 0; j < 8; ++j) {
        int r = tile * 16 + roff + j;
        hout[r * HH +  0 + m] = acc0[j];
        hout[r * HH + 16 + m] = acc1[j];
        hout[r * HH + 32 + m] = acc2[j];
        hout[r * HH + 48 + m] = acc3[j];
    }
}

// ---------------------------------------------------------------------
// head: logits = h @ W_out^T  (64->2) + log_softmax
// ---------------------------------------------------------------------
__global__ void __launch_bounds__(256) head_kernel(const float* __restrict__ h,
                                                   const float* __restrict__ Wo,
                                                   float* __restrict__ out) {
    int n = blockIdx.x * blockDim.x + threadIdx.x;
    if (n >= NN) return;
    float l0 = 0.f, l1 = 0.f;
#pragma unroll 8
    for (int k = 0; k < HH; ++k) {
        float v = h[n * HH + k];
        l0 += v * Wo[k];
        l1 += v * Wo[HH + k];
    }
    float mx  = fmaxf(l0, l1);
    float lse = mx + __logf(__expf(l0 - mx) + __expf(l1 - mx));
    out[n * 2 + 0] = l0 - lse;
    out[n * 2 + 1] = l1 - lse;
}

// ---------------------------------------------------------------------
extern "C" void kernel_launch(void* const* d_in, const int* in_sizes, int n_in,
                              void* d_out, int out_size, void* d_ws, size_t ws_size,
                              hipStream_t stream) {
    const float* x      = (const float*)d_in[0];
    const int*   ei     = (const int*)  d_in[1];
    const float* ew     = (const float*)d_in[2];
    const float* Win    = (const float*)d_in[3];
    const float* coeffs = (const float*)d_in[4];
    const float* Wout   = (const float*)d_in[5];
    float* out = (float*)d_out;

    char* ws = (char*)d_ws;
    size_t off = 0;
    auto carve = [&](size_t bytes) -> void* {
        void* p = ws + off;
        off = (off + bytes + 255) & ~(size_t)255;
        return p;
    };
    float*     hA = (float*)    carve((size_t)NN * HH * sizeof(float));
    float*     hB = (float*)    carve((size_t)NN * HH * sizeof(float));
    _Float16*  Wb = (_Float16*) carve((size_t)4 * 6 * 32 * 16 * sizeof(_Float16));
    _Float16*  Bp = (_Float16*) carve((size_t)NKAN * 4 * 32 * 32 * 16 * sizeof(_Float16));
    (void)ws_size; (void)in_sizes; (void)n_in; (void)out_size;

    // 1) repack weights into WMMA B-fragment layouts
    prep_win<<<(4 * 6 * 32 * 16 + 255) / 256, 256, 0, stream>>>(Win, Wb);
    prep_bp <<<(NKAN * 4 * 32 * 32 * 16 + 255) / 256, 256, 0, stream>>>(coeffs, Bp);

    // 2) h = x @ W_in^T
    const int tile_blocks = (NTILES + WAVES_PER_BLK - 1) / WAVES_PER_BLK;
    lin_in_kernel<<<tile_blocks, 256, 0, stream>>>(x, Wb, hA);

    // 3) two (SpMM -> KAN) layers, ping-ponging hA/hB
    const unsigned spmm_threads = (unsigned)EE * HH;
    const unsigned spmm_blocks  = (spmm_threads + 255u) / 256u;
    for (int l = 0; l < NKAN; ++l) {
        hipMemsetAsync(hB, 0, (size_t)NN * HH * sizeof(float), stream);
        spmm_kernel<<<spmm_blocks, 256, 0, stream>>>(ei, ew, hA, hB);
        kan_kernel<<<tile_blocks, 256, 0, stream>>>(hB, Bp + (size_t)l * 4 * 32 * 32 * 16, hA);
    }

    // 4) logits + log_softmax
    head_kernel<<<(NN + 255) / 256, 256, 0, stream>>>(hA, Wout, out);
}
